// MultiHeadSelfAttention_21586505629867
// MI455X (gfx1250) — matrix operand order repaired and checked by reading, not compiled, not run on hardware
//
#include <hip/hip_runtime.h>
#include <hip/hip_bf16.h>
#include <math.h>

// ---------------------------------------------------------------------------
// MHSA for MI455X (gfx1250, wave32): bf16 WMMA everywhere, f32 accumulate.
//   B=2, T=2048, D=768, H=12, Dh=64.
// Pipeline:
//   1) convert x, Wq/Wk/Wv/Wo -> bf16 (workspace)
//   2) GEMM (wmma bf16): Q=[B,H,T,64], K=[B,H,T,64], V stored transposed [B,H,64,T]
//   3) flash-attention (wmma bf16, online softmax) -> ctx bf16 [B,T,768]
//   4) GEMM (wmma bf16): out = ctx @ Wo^T + bo  (f32 out)
// Round 2: GEMM register blocking raised to 32(M)x64(N) per wave
//          (8 accumulators, 2 A-frags x 4 shared B-frags -> 1.5 loads/WMMA).
// ---------------------------------------------------------------------------

typedef __attribute__((ext_vector_type(16))) __bf16 v16bf;
typedef __attribute__((ext_vector_type(8)))  __bf16 v8bf;
typedef __attribute__((ext_vector_type(8)))  float  v8f;

#define BATCH   2
#define SEQ     2048
#define DIM     768
#define HEADS   12
#define DHEAD   64
#define MROWS   (BATCH * SEQ)          // 4096

// Load a v16bf fragment from two 16-byte-aligned 8-element chunks.
__device__ __forceinline__ v16bf ldpair(const __bf16* p0, const __bf16* p1) {
    v8bf lo = *(const v8bf*)p0;
    v8bf hi = *(const v8bf*)p1;
    return __builtin_shufflevector(lo, hi, 0,1,2,3,4,5,6,7,8,9,10,11,12,13,14,15);
}

__device__ __forceinline__ v8f wmma_bf16(v16bf a, v16bf b, v8f c) {
    return __builtin_amdgcn_wmma_f32_16x16x32_bf16(false, a, false, b, (short)0, c,
                                                   false, false);
}

// --------------------------- f32 -> bf16 convert ---------------------------
__global__ void cvt_f32_bf16(const float* __restrict__ in, __bf16* __restrict__ out,
                             int n) {
    int i = blockIdx.x * blockDim.x + threadIdx.x;
    if (i < n) out[i] = (__bf16)in[i];
}

// ------------------------------- GEMM stage --------------------------------
// y[m,n] = sum_k A[m,k] * W[n,k] + bias[n];  M=4096, N=768, K=768.
// block = 256 threads (8 waves); wave computes 32(M) x 64(N); block 256x64.
// mode 0: out bf16 [B,H,T,64]      (Q, K)
// mode 1: out bf16 [B,H,64,T]      (V, pre-transposed for PV B-fragments)
// mode 2: out f32  [B,T,768]       (final projection)
__global__ __launch_bounds__(256)
void gemm_bf16(const __bf16* __restrict__ A, const __bf16* __restrict__ W,
               const float* __restrict__ bias, void* __restrict__ outp, int mode) {
    const int lane = threadIdx.x & 31;
    const int wid  = threadIdx.x >> 5;
    const int g    = lane >> 4;      // khalf / row-group select
    const int ln   = lane & 15;
    const int m0   = blockIdx.x * 256 + wid * 32;
    const int n0   = blockIdx.y * 64;

    const v8f zero = {};
    v8f acc[2][4] = {{zero, zero, zero, zero}, {zero, zero, zero, zero}};

    const __bf16* arow0 = A + (size_t)(m0 + ln) * DIM;
    const __bf16* arow1 = arow0 + (size_t)16 * DIM;
    for (int k0 = 0; k0 < DIM; k0 += 32) {
        // A fragments: K = 8*g + e (e<8), 16 + 8*g + (e-8) (e>=8)
        v16bf af0 = ldpair(arow0 + k0 + 8 * g, arow0 + k0 + 16 + 8 * g);
        v16bf af1 = ldpair(arow1 + k0 + 8 * g, arow1 + k0 + 16 + 8 * g);
#pragma unroll
        for (int nt = 0; nt < 4; ++nt) {
            // B fragment: lane holds column n = row n of W, K contiguous.
            const __bf16* wrow = W + (size_t)(n0 + 16 * nt + ln) * DIM + k0 + 8 * g;
            v16bf bf = ldpair(wrow, wrow + 16);
            acc[0][nt] = wmma_bf16(af0, bf, acc[0][nt]);
            acc[1][nt] = wmma_bf16(af1, bf, acc[1][nt]);
        }
    }

#pragma unroll
    for (int mt = 0; mt < 2; ++mt) {
#pragma unroll
        for (int nt = 0; nt < 4; ++nt) {
            const int n    = n0 + 16 * nt + ln;
            const float bn = bias[n];
#pragma unroll
            for (int r = 0; r < 8; ++r) {
                const int m = m0 + 16 * mt + 8 * g + r;  // C/D: M = r + 8*(lane>>4)
                const float v = acc[mt][nt][r] + bn;
                if (mode == 2) {
                    ((float*)outp)[(size_t)m * DIM + n] = v;
                } else {
                    const int b = m >> 11, t = m & (SEQ - 1);
                    const int h = n >> 6,  d = n & (DHEAD - 1);
                    __bf16* o = (__bf16*)outp;
                    if (mode == 0)
                        o[((size_t)(b * HEADS + h) * SEQ + t) * DHEAD + d] = (__bf16)v;
                    else
                        o[((size_t)(b * HEADS + h) * DHEAD + d) * SEQ + t] = (__bf16)v;
                }
            }
        }
    }
}

// --------------------------- flash attention -------------------------------
// One wave handles 16 query rows of one (b,h). Keys streamed in blocks of 32.
// Q,K: bf16 [B,H,T,64]; Vt: bf16 [B,H,64,T]; ctx out: bf16 [B,T,768].
#define PROW 40   // LDS row stride (shorts) to spread banks; 80B keeps 16B align

__global__ __launch_bounds__(256)
void attn_flash(const __bf16* __restrict__ Q, const __bf16* __restrict__ K,
                const __bf16* __restrict__ Vt, __bf16* __restrict__ ctx) {
    __shared__ __bf16 pbuf[8][16 * PROW];

    const int lane = threadIdx.x & 31;
    const int wid  = threadIdx.x >> 5;
    const int g    = lane >> 4;
    const int ln   = lane & 15;

    const int task = blockIdx.x * 8 + wid;        // 3072 tasks
    const int qt   = task & 127;                  // query tile (16 rows each)
    const int hh   = (task >> 7) % HEADS;
    const int bb   = task / (128 * HEADS);
    const size_t bh = (size_t)bb * HEADS + hh;

    const __bf16* Qb = Q  + bh * SEQ * DHEAD;
    const __bf16* Kb = K  + bh * SEQ * DHEAD;
    const __bf16* Vb = Vt + bh * DHEAD * SEQ;

    // Q A-fragments for d=0..31 and d=32..63; fold in 1/sqrt(64) once.
    const __bf16* qrow = Qb + (size_t)(qt * 16 + ln) * DHEAD;
    v16bf qa0 = ldpair(qrow + 8 * g,      qrow + 16 + 8 * g);
    v16bf qa1 = ldpair(qrow + 32 + 8 * g, qrow + 48 + 8 * g);
#pragma unroll
    for (int i = 0; i < 16; ++i) {
        qa0[i] = (__bf16)((float)qa0[i] * 0.125f);
        qa1[i] = (__bf16)((float)qa1[i] * 0.125f);
    }

    const v8f zero = {};
    v8f O[4] = {zero, zero, zero, zero};
    float mrow[8], lrow[8];
#pragma unroll
    for (int r = 0; r < 8; ++r) { mrow[r] = -__builtin_inff(); lrow[r] = 0.0f; }

    __bf16* pw = &pbuf[wid][0];

    for (int kbase = 0; kbase < SEQ; kbase += 32) {
        // ----- scores S = (Q/8) @ K^T, two 16-key tiles, K-dim = Dh = 64 -----
        v8f S[2];
#pragma unroll
        for (int nt = 0; nt < 2; ++nt) {
            const __bf16* krow = Kb + (size_t)(kbase + 16 * nt + ln) * DHEAD;
            v16bf kf0 = ldpair(krow + 8 * g,      krow + 16 + 8 * g);
            v16bf kf1 = ldpair(krow + 32 + 8 * g, krow + 48 + 8 * g);
            v8f s = zero;
            s = wmma_bf16(qa0, kf0, s);
            s = wmma_bf16(qa1, kf1, s);
            S[nt] = s;
        }

        // ----- online softmax (rows live across 16 lanes of a group) -----
        float mnew[8], corr[8];
#pragma unroll
        for (int r = 0; r < 8; ++r) {
            float v = fmaxf(S[0][r], S[1][r]);
            v = fmaxf(v, __shfl_xor(v, 1, 32));
            v = fmaxf(v, __shfl_xor(v, 2, 32));
            v = fmaxf(v, __shfl_xor(v, 4, 32));
            v = fmaxf(v, __shfl_xor(v, 8, 32));
            const float mn = fmaxf(mrow[r], v);
            corr[r] = __expf(mrow[r] - mn);
            mnew[r] = mn;
            mrow[r] = mn;
        }
#pragma unroll
        for (int dt = 0; dt < 4; ++dt)
#pragma unroll
            for (int r = 0; r < 8; ++r) O[dt][r] *= corr[r];

        // P = exp(S - m); row sums; stage P to LDS as bf16 (D-layout write)
#pragma unroll
        for (int r = 0; r < 8; ++r) {
            const float p0 = __expf(S[0][r] - mnew[r]);
            const float p1 = __expf(S[1][r] - mnew[r]);
            pw[(8 * g + r) * PROW + ln]      = (__bf16)p0;
            pw[(8 * g + r) * PROW + 16 + ln] = (__bf16)p1;
            float rs = p0 + p1;
            rs += __shfl_xor(rs, 1, 32);
            rs += __shfl_xor(rs, 2, 32);
            rs += __shfl_xor(rs, 4, 32);
            rs += __shfl_xor(rs, 8, 32);
            lrow[r] = lrow[r] * corr[r] + rs;
        }
        __syncthreads();   // order LDS write->read; keeps waves cache-phased

        // Re-read P as an A-fragment (16x32), then O += P @ V
        v16bf pa = ldpair(pw + ln * PROW + 8 * g, pw + ln * PROW + 16 + 8 * g);
#pragma unroll
        for (int dt = 0; dt < 4; ++dt) {
            const __bf16* vrow = Vb + (size_t)(16 * dt + ln) * SEQ + kbase + 8 * g;
            v16bf vf = ldpair(vrow, vrow + 16);
            O[dt] = wmma_bf16(pa, vf, O[dt]);
        }
        __syncthreads();
    }

    // ----- normalize and write ctx [B,T,768] -----
    float il[8];
#pragma unroll
    for (int r = 0; r < 8; ++r) il[r] = 1.0f / lrow[r];
#pragma unroll
    for (int dt = 0; dt < 4; ++dt)
#pragma unroll
        for (int r = 0; r < 8; ++r) {
            const int t   = qt * 16 + 8 * g + r;
            const int col = hh * DHEAD + 16 * dt + ln;
            ctx[((size_t)bb * SEQ + t) * DIM + col] = (__bf16)(O[dt][r] * il[r]);
        }
}

// ------------------------------ launch glue --------------------------------
extern "C" void kernel_launch(void* const* d_in, const int* in_sizes, int n_in,
                              void* d_out, int out_size, void* d_ws, size_t ws_size,
                              hipStream_t stream) {
    (void)in_sizes; (void)n_in; (void)out_size; (void)ws_size;

    const float* x  = (const float*)d_in[0];
    const float* Wq = (const float*)d_in[1];
    const float* bq = (const float*)d_in[2];
    const float* Wk = (const float*)d_in[3];
    const float* bk = (const float*)d_in[4];
    const float* Wv = (const float*)d_in[5];
    const float* bv = (const float*)d_in[6];
    const float* Wo = (const float*)d_in[7];
    const float* bo = (const float*)d_in[8];

    // workspace carve-up (bf16 elements), ~36.2 MB total
    __bf16* xb   = (__bf16*)d_ws;                 // 4096*768
    __bf16* wqb  = xb  + (size_t)MROWS * DIM;
    __bf16* wkb  = wqb + (size_t)DIM * DIM;
    __bf16* wvb  = wkb + (size_t)DIM * DIM;
    __bf16* wob  = wvb + (size_t)DIM * DIM;
    __bf16* qb   = wob + (size_t)DIM * DIM;       // [B,H,T,64]
    __bf16* kb   = qb  + (size_t)MROWS * DIM;
    __bf16* vtb  = kb  + (size_t)MROWS * DIM;     // [B,H,64,T]
    __bf16* ctxb = vtb + (size_t)MROWS * DIM;     // [B,T,768]

    const int nx = MROWS * DIM;       // 3,145,728
    const int nw = DIM * DIM;         //   589,824
    cvt_f32_bf16<<<dim3((nx + 255) / 256), 256, 0, stream>>>(x,  xb,  nx);
    cvt_f32_bf16<<<dim3((nw + 255) / 256), 256, 0, stream>>>(Wq, wqb, nw);
    cvt_f32_bf16<<<dim3((nw + 255) / 256), 256, 0, stream>>>(Wk, wkb, nw);
    cvt_f32_bf16<<<dim3((nw + 255) / 256), 256, 0, stream>>>(Wv, wvb, nw);
    cvt_f32_bf16<<<dim3((nw + 255) / 256), 256, 0, stream>>>(Wo, wob, nw);

    dim3 gg(MROWS / 256, DIM / 64);   // (16, 12)
    gemm_bf16<<<gg, 256, 0, stream>>>(xb, wqb, bq, (void*)qb,  0);
    gemm_bf16<<<gg, 256, 0, stream>>>(xb, wkb, bk, (void*)kb,  0);
    gemm_bf16<<<gg, 256, 0, stream>>>(xb, wvb, bv, (void*)vtb, 1);

    attn_flash<<<dim3(BATCH * HEADS * (SEQ / 16) / 8), 256, 0, stream>>>(qb, kb, vtb,
                                                                         ctxb);

    gemm_bf16<<<gg, 256, 0, stream>>>(ctxb, wob, bo, d_out, 2);
}
